// MoE_20298015441100
// MI455X (gfx1250) — matrix operand order repaired and checked by reading, compiled, not run on hardware
//
#include <hip/hip_runtime.h>
#include <hip/hip_bf16.h>

// ---------------------------------------------------------------------------
// MoE (sigmoid top-4 of 32 experts + shared expert) for gfx1250 / MI455X.
//   1) gate_topk: wave32-per-token gating, per-expert token lists
//   2) swizzle_w: one-time f32 -> bf16 convert + WMMA-B-layout swizzle (in ws)
//   3) moe_ffn:   grouped SwiGLU GEMMs, bf16 WMMA, B frags straight from L2
// ---------------------------------------------------------------------------

typedef __attribute__((ext_vector_type(16))) __bf16 v16bf;
typedef __attribute__((ext_vector_type(8)))  float  v8f;

#define MOE_N   2048
#define MOE_C   1024
#define MOE_E   32
#define MOE_K   4

// f32 -> bf16 (round to nearest even; NaN edge ignored)
static __device__ __forceinline__ unsigned short f2bf(float f) {
  unsigned u = __float_as_uint(f);
  u += 0x7FFFu + ((u >> 16) & 1u);
  return (unsigned short)(u >> 16);
}

// sigmoid with v_rcp_f32 (1 ulp) instead of IEEE division chain
static __device__ __forceinline__ float fast_sigmoid(float v) {
  return __builtin_amdgcn_rcpf(1.f + __expf(-v));
}

static __device__ __forceinline__ v8f zero8() {
  v8f z;
#pragma unroll
  for (int i = 0; i < 8; ++i) z[i] = 0.f;
  return z;
}

union BF16Frag {
  v16bf  v;
  float4 f4[2];
};

// A-matrix 16x32 bf16 fragment from row-major LDS [16][stride] (half units).
// ISA: lanes 0-15: M=lane, K=0..7 & 16..23 ; lanes 16-31: M=lane-16, K=8..15 & 24..31.
static __device__ __forceinline__ v16bf load_a_frag(const unsigned short* base,
                                                    int stride, int kOff, int lane) {
  int m   = lane & 15;
  int kb8 = (lane >> 4) << 3;
  const unsigned short* p = base + m * stride + kOff + kb8;
  BF16Frag u;
  u.f4[0] = *(const float4*)(const void*)(p);
  u.f4[1] = *(const float4*)(const void*)(p + 16);
  return u.v;
}

// B-matrix 32x16 bf16 fragment from the pre-swizzled global layout:
// slab kb contiguous, element (kl, col) at [(col * 32) + kl] halves.
// ISA: lanes 0-15: N=lane, K=0..7 & 16..23 ; lanes 16-31: N=lane-16, K=8..15 & 24..31.
// -> two 8-half runs per lane (at kh, kh+16): two global_load_b128, wave covers 1KB.
static __device__ __forceinline__ v16bf load_b_sw(const unsigned short* slab,
                                                  int colBase, int lane) {
  int n  = lane & 15;
  int kh = (lane >> 4) << 3;
  const unsigned short* p = slab + (size_t)(colBase + n) * 32 + kh;
  BF16Frag u;
  u.f4[0] = *(const float4*)(const void*)(p);
  u.f4[1] = *(const float4*)(const void*)(p + 16);
  return u.v;
}

static __device__ __forceinline__ v8f wmma_bf16(v16bf a, v16bf b, v8f c) {
  return __builtin_amdgcn_wmma_f32_16x16x32_bf16(
      false, a, false, b, (short)0, c, false, false);
}

// ---------------------------------------------------------------------------
// Kernel 1: gating. One wave32 per token; lane == expert (E == 32).
// ---------------------------------------------------------------------------
__global__ __launch_bounds__(256) void gate_topk(
    const float* __restrict__ x, const float* __restrict__ gw,
    const float* __restrict__ gbias, int* __restrict__ cnt,
    int* __restrict__ tok_idx, float* __restrict__ tok_w) {
  int wave = threadIdx.x >> 5;
  int lane = threadIdx.x & 31;
  int n = blockIdx.x * 8 + wave;

  const float4* xv = (const float4*)(const void*)(x + (size_t)n * MOE_C);
  float acc = 0.f;
#pragma unroll 4
  for (int c4 = 0; c4 < MOE_C / 4; ++c4) {
    float4 xx = xv[c4];
    int cb = c4 * 4;
    acc += xx.x * gw[(cb + 0) * MOE_E + lane];
    acc += xx.y * gw[(cb + 1) * MOE_E + lane];
    acc += xx.z * gw[(cb + 2) * MOE_E + lane];
    acc += xx.w * gw[(cb + 3) * MOE_E + lane];
  }
  float s    = fast_sigmoid(acc);
  float cand = s + gbias[lane];

  float vk[MOE_K];
  int   ik[MOE_K];
#pragma unroll
  for (int it = 0; it < MOE_K; ++it) {
    unsigned ub = __float_as_uint(cand);
    ub ^= (ub >> 31) ? 0xFFFFFFFFu : 0x80000000u;  // monotone float -> uint
    unsigned long long key = ((unsigned long long)ub << 6) | (unsigned)lane;
#pragma unroll
    for (int off = 16; off >= 1; off >>= 1) {
      unsigned long long o = __shfl_xor(key, off, 32);
      if (o > key) key = o;
    }
    int sel = (int)(key & 63ull);
    ik[it]  = sel;
    vk[it]  = __shfl(s, sel, 32);
    if (lane == sel) cand = -3.0e38f;
  }
  float wsum = vk[0] + vk[1] + vk[2] + vk[3];

  if (lane < MOE_K) {
    int   e = (lane == 0) ? ik[0] : (lane == 1) ? ik[1] : (lane == 2) ? ik[2] : ik[3];
    float v = (lane == 0) ? vk[0] : (lane == 1) ? vk[1] : (lane == 2) ? vk[2] : vk[3];
    float w = (v / wsum) * 1.0f;  // ROUTE_SCALE
    int slot = atomicAdd(cnt + e, 1);
    tok_idx[e * MOE_N + slot] = n;
    tok_w[e * MOE_N + slot]   = w;
  }
}

// ---------------------------------------------------------------------------
// Kernel 2: weight convert + swizzle into WMMA-B-ready bf16 layout.
// src: (batch, K, NCOL) f32 row-major ; dst: per batch, per 32-row k-slab,
// column-major [col][32] halves, slabs contiguous. blockIdx.y = batch.
// Each thread packs 2 consecutive K values of one column -> one b32 store.
// ---------------------------------------------------------------------------
template <int K, int NCOL>
__global__ __launch_bounds__(256) void swizzle_w(const float* __restrict__ src,
                                                 unsigned short* __restrict__ dst) {
  static_assert(NCOL == 1024 || NCOL == 2048, "pow2 cols");
  constexpr int LOGN = (NCOL == 2048) ? 11 : 10;
  size_t base = (size_t)blockIdx.y * K * NCOL;
  int idx = blockIdx.x * 256 + threadIdx.x;  // pair index within batch
  int col = idx & (NCOL - 1);
  int r   = idx >> LOGN;
  int klp = r & 15;          // kl pair (kl = 2*klp, 2*klp+1)
  int kb  = r >> 4;          // k-slab
  const float* s = src + base + (size_t)(kb * 32 + 2 * klp) * NCOL + col;
  float f0 = s[0];
  float f1 = s[NCOL];
  unsigned pk = (unsigned)f2bf(f0) | ((unsigned)f2bf(f1) << 16);
  unsigned* d = (unsigned*)(dst + base);
  d[((size_t)kb * NCOL + col) * 16 + klp] = pk;
}

// ---------------------------------------------------------------------------
// Kernel 3: SwiGLU FFN over an M = MTILES*16 token tile. 8 waves / block.
//   GEMM1: (M x C) @ (C x 2H) fused SiLU*y -> bf16 act in LDS
//   GEMM2: (M x H) @ (H x C)  -> scaled store/atomicAdd
// N dimension processed in chunks (1 N-tile per wave) over full K, so the
// only barriers are after x staging and before GEMM2.
// Weights are the pre-swizzled bf16 layout; e0 offsets the expert id when the
// weight buffer holds only a sub-batch (low-workspace fallback).
// ---------------------------------------------------------------------------
template <int HID, int MTILES, bool GATHER>
__global__ __launch_bounds__(256) void moe_ffn(
    const float*          __restrict__ x,
    const unsigned short* __restrict__ wg,   // bf16 swizzled (batch, C, 2H)
    const unsigned short* __restrict__ wd,   // bf16 swizzled (batch, H, C)
    const int*            __restrict__ tok_idx,
    const float*          __restrict__ tok_w,
    const int*            __restrict__ cnt,
    float*                __restrict__ out,
    int e0) {
  constexpr int C      = MOE_C;
  constexpr int TWO_H  = 2 * HID;
  constexpr int M      = MTILES * 16;
  constexpr int PTOT   = HID / 16;        // (y,g) pair tiles in GEMM1
  constexpr int CH1    = PTOT / 8;        // GEMM1 N-chunks
  constexpr int CH2    = (C / 16) / 8;    // GEMM2 N-chunks (= 8)
  constexpr int KS1    = C / 32;
  constexpr int KS2    = HID / 32;
  constexpr int TILES  = MOE_N / M;

  extern __shared__ char smem_raw[];
  unsigned short* lds_x   = (unsigned short*)smem_raw;   // [M][C]
  unsigned short* lds_act = lds_x + M * C;               // [M][HID]
  float* lds_wt  = (float*)(lds_act + (size_t)M * HID);  // [M]
  int*   lds_tok = (int*)(lds_wt + M);                   // [M]

  int e = 0, eb = 0, tile, nrows;
  if (GATHER) {
    eb    = blockIdx.x / TILES;   // batch index into the weight buffer
    e     = e0 + eb;              // absolute expert id
    tile  = blockIdx.x % TILES;
    nrows = cnt[e];
    if (tile * M >= nrows) return;
  } else {
    tile  = blockIdx.x;
    nrows = M;
  }

  const unsigned short* wgS = wg + (size_t)eb * C * TWO_H;
  const unsigned short* wdS = wd + (size_t)eb * HID * C;

  const int tid  = threadIdx.x;
  const int wave = tid >> 5;
  const int lane = tid & 31;

  if (tid < M) {
    if (GATHER) {
      int  slot = tile * M + tid;
      bool val  = slot < nrows;
      lds_tok[tid] = val ? tok_idx[e * MOE_N + slot] : 0;
      lds_wt[tid]  = val ? tok_w[e * MOE_N + slot] : 0.f;
    } else {
      lds_tok[tid] = tile * M + tid;
      lds_wt[tid]  = 1.f;
    }
  }
  __syncthreads();

  // stage M token rows of x as bf16 (coalesced over columns)
#pragma unroll
  for (int i = 0; i < (M * C) / 256; ++i) {
    int idx = tid + i * 256;
    int m   = idx >> 10;  // / C
    int c   = idx & (C - 1);
    lds_x[m * C + c] = f2bf(x[(size_t)lds_tok[m] * C + c]);
  }
  __syncthreads();

  const int mB  = (lane >> 4) * 8;
  const int nLn = lane & 15;

  // ---------------- GEMM1 (chunked over N pairs) + fused SiLU --------------
  for (int ch = 0; ch < CH1; ++ch) {
    v8f accY[MTILES], accG[MTILES];
#pragma unroll
    for (int mt = 0; mt < MTILES; ++mt) { accY[mt] = zero8(); accG[mt] = zero8(); }

    const int jy = ch * 8 + wave;
    const int cy = jy * 16;
    const int cg = cy + HID;

    for (int kb = 0; kb < KS1; ++kb) {
      const unsigned short* slab = wgS + (size_t)kb * TWO_H * 32;
      v16bf by = load_b_sw(slab, cy, lane);
      v16bf bg = load_b_sw(slab, cg, lane);
      if (lane == 0 && kb + 1 < KS1)  // warm L2 for the next slab
        __builtin_prefetch(slab + (size_t)TWO_H * 32, 0, 1);
#pragma unroll
      for (int mt = 0; mt < MTILES; ++mt) {
        v16bf a = load_a_frag(lds_x + mt * 16 * C, C, kb * 32, lane);
        accY[mt] = wmma_bf16(a, by, accY[mt]);
        accG[mt] = wmma_bf16(a, bg, accG[mt]);
      }
    }

    // act = silu(g) * y -> bf16 LDS (disjoint columns per wave/chunk)
    const int colA = cy + nLn;
#pragma unroll
    for (int mt = 0; mt < MTILES; ++mt) {
#pragma unroll
      for (int r = 0; r < 8; ++r) {
        float y = accY[mt][r];
        float g = accG[mt][r];
        float a = g * fast_sigmoid(g) * y;   // silu(g) * y, rcp-based
        lds_act[(size_t)(mt * 16 + mB + r) * HID + colA] = f2bf(a);
      }
    }
  }
  __syncthreads();

  // ---------------- GEMM2 (chunked over N tiles) ---------------------------
  for (int ch = 0; ch < CH2; ++ch) {
    v8f acc[MTILES];
#pragma unroll
    for (int mt = 0; mt < MTILES; ++mt) acc[mt] = zero8();

    const int j    = ch * 8 + wave;
    const int colB = j * 16;

    for (int kb = 0; kb < KS2; ++kb) {
      v16bf b = load_b_sw(wdS + (size_t)kb * C * 32, colB, lane);
#pragma unroll
      for (int mt = 0; mt < MTILES; ++mt) {
        v16bf a = load_a_frag(lds_act + mt * 16 * HID, HID, kb * 32, lane);
        acc[mt] = wmma_bf16(a, b, acc[mt]);
      }
    }

    const int col = colB + nLn;
#pragma unroll
    for (int mt = 0; mt < MTILES; ++mt) {
#pragma unroll
      for (int r = 0; r < 8; ++r) {
        int   m   = mt * 16 + mB + r;
        float val = acc[mt][r] * lds_wt[m];
        if (GATHER) {
          if (tile * M + m < nrows)
            atomicAdd(out + (size_t)lds_tok[m] * C + col, val);
        } else {
          out[(size_t)(tile * M + m) * C + col] = val;
        }
      }
    }
  }
}

// ---------------------------------------------------------------------------
// Launch
// ---------------------------------------------------------------------------
extern "C" void kernel_launch(void* const* d_in, const int* in_sizes, int n_in,
                              void* d_out, int out_size, void* d_ws, size_t ws_size,
                              hipStream_t stream) {
  (void)in_sizes; (void)n_in; (void)out_size;

  const float* x             = (const float*)d_in[0];  // (N, C)
  const float* gate_w        = (const float*)d_in[1];  // (C, E)
  const float* gate_bias     = (const float*)d_in[2];  // (E)
  const float* shared_gate_w = (const float*)d_in[3];  // (C, 2048)
  const float* shared_down_w = (const float*)d_in[4];  // (1024, C)
  const float* expert_gate_w = (const float*)d_in[5];  // (E, C, 1024)
  const float* expert_down_w = (const float*)d_in[6];  // (E, 512, C)
  float* out = (float*)d_out;                          // (N, C)

  // workspace layout
  char* ws = (char*)d_ws;
  int*   cnt     = (int*)ws;
  int*   tok_idx = (int*)(ws + 256);
  float* tok_w   = (float*)(ws + 256 + (size_t)MOE_E * MOE_N * 4);
  const size_t off_lists = 256 + 2 * (size_t)MOE_E * MOE_N * 4;

  size_t off = off_lists;
  unsigned short* swz_eg = (unsigned short*)(ws + off);  off += (size_t)32 * 1024 * 1024 * 2;
  unsigned short* swz_ed = (unsigned short*)(ws + off);  off += (size_t)32 * 512  * 1024 * 2;
  unsigned short* swz_sg = (unsigned short*)(ws + off);  off += (size_t)1024 * 2048 * 2;
  unsigned short* swz_sd = (unsigned short*)(ws + off);  off += (size_t)1024 * 1024 * 2;
  const bool full = ws_size >= off;  // ~103 MB for the one-shot prepass

  hipMemsetAsync(cnt, 0, MOE_E * sizeof(int), stream);

  // 1) gating + per-expert token lists
  gate_topk<<<MOE_N / 8, 256, 0, stream>>>(x, gate_w, gate_bias, cnt, tok_idx, tok_w);

  const size_t smemS = ((size_t)32 * MOE_C + 32 * 1024) * 2 + 32 * 8;  // M=32, H=1024
  const size_t smemR = ((size_t)64 * MOE_C + 64 * 512)  * 2 + 64 * 8;  // M=64, H=512

  if (full) {
    // 2) one-shot weight convert + swizzle (memory bound, ~103 MB written)
    swizzle_w<1024, 1024><<<dim3((1024 * 1024 / 2) / 256, MOE_E), 256, 0, stream>>>(expert_gate_w, swz_eg);
    swizzle_w<512,  1024><<<dim3((512  * 1024 / 2) / 256, MOE_E), 256, 0, stream>>>(expert_down_w, swz_ed);
    swizzle_w<1024, 2048><<<dim3((1024 * 2048 / 2) / 256, 1),     256, 0, stream>>>(shared_gate_w, swz_sg);
    swizzle_w<1024, 1024><<<dim3((1024 * 1024 / 2) / 256, 1),     256, 0, stream>>>(shared_down_w, swz_sd);

    // 3) shared expert initializes out; 4) routed experts atomically add
    moe_ffn<1024, 2, false><<<MOE_N / 32, 256, smemS, stream>>>(
        x, swz_sg, swz_sd, nullptr, nullptr, nullptr, out, 0);
    moe_ffn<512, 4, true><<<MOE_E * (MOE_N / 64), 256, smemR, stream>>>(
        x, swz_eg, swz_ed, tok_idx, tok_w, cnt, out, 0);
  } else {
    // low-workspace fallback: swizzle into a small reused buffer, expert by
    // expert (stream ordering serializes swizzle -> ffn -> next swizzle).
    unsigned short* buf  = (unsigned short*)(ws + off_lists);
    unsigned short* buf2 = buf + (size_t)1024 * 2048;  // second matrix slot

    swizzle_w<1024, 2048><<<dim3((1024 * 2048 / 2) / 256, 1), 256, 0, stream>>>(shared_gate_w, buf);
    swizzle_w<1024, 1024><<<dim3((1024 * 1024 / 2) / 256, 1), 256, 0, stream>>>(shared_down_w, buf2);
    moe_ffn<1024, 2, false><<<MOE_N / 32, 256, smemS, stream>>>(
        x, buf, buf2, nullptr, nullptr, nullptr, out, 0);

    for (int e = 0; e < MOE_E; ++e) {
      swizzle_w<1024, 1024><<<dim3((1024 * 1024 / 2) / 256, 1), 256, 0, stream>>>(
          expert_gate_w + (size_t)e * 1024 * 1024, buf);
      swizzle_w<512, 1024><<<dim3((512 * 1024 / 2) / 256, 1), 256, 0, stream>>>(
          expert_down_w + (size_t)e * 512 * 1024, buf2);
      moe_ffn<512, 4, true><<<MOE_N / 64, 256, smemR, stream>>>(
          x, buf, buf2, tok_idx, tok_w, cnt, out, e);
    }
  }
}
